// LatentODEVAE_44049184587813
// MI455X (gfx1250) — matrix operand order repaired and checked by reading, not compiled, not run on hardware
//
#include <hip/hip_runtime.h>

// ---------------- problem constants ----------------
#define OBS   2
#define LAT   4
#define RNN   25
#define ODEH  20
#define BATCH 4096
#define SEQL  200
#define TT    500
#define GN    75      // 3*RNN gate width
#define GNP   80      // padded to 5 tiles of 16
#define NT    5       // N tiles of 16
#define HSTR  33      // padded row stride for h in LDS (cols 25..31 stay zero)
#define GSTR  20      // padded col stride for gate dumps (col-major [80][GSTR])
#define LOG2E 1.4426950408889634f

typedef __attribute__((ext_vector_type(16))) _Float16 v16h;
typedef __attribute__((ext_vector_type(8)))  float    v8f;

// Raw-hardware transcendentals: v_rcp_f32 / v_exp_f32, no IEEE-div expansion.
__device__ __forceinline__ float rcp_hw(float x)  { return __builtin_amdgcn_rcpf(x); }
__device__ __forceinline__ float exp2_hw(float x) { return __builtin_amdgcn_exp2f(x); }
__device__ __forceinline__ float sigm_f(float x) {
  return rcp_hw(1.0f + exp2_hw(-LOG2E * x));
}
__device__ __forceinline__ float tanh_ff(float x) {
  // tanh(x) = 1 - 2/(exp(2x)+1)
  return 1.0f - 2.0f * rcp_hw(1.0f + exp2_hw((2.0f * LOG2E) * x));
}

// =====================================================================
// Kernel 1: reverse-time GRU encoder, WMMA 16x16x32 f16 (f32 accumulate)
// One wave per 16-row batch tile; weights register-resident in B-layout.
// Branch-free inner loops (selects only) to avoid saveexec churn.
// =====================================================================
__global__ __launch_bounds__(32) void gru_encoder_wmma(
    const float* __restrict__ obs,      // [B, L, OBS]
    const float* __restrict__ enc_in_W, // [2, 25]
    const float* __restrict__ enc_in_b, // [25]
    const float* __restrict__ Wih,      // [25, 75]
    const float* __restrict__ Whh,      // [25, 75]
    const float* __restrict__ bih,      // [75]
    const float* __restrict__ bhh,      // [75]
    const float* __restrict__ statsW,   // [25, 8]
    const float* __restrict__ statsb,   // [8]
    float* __restrict__ out_z0,         // [B, 4]
    float* __restrict__ out_mu,         // [B, 4]
    float* __restrict__ out_lv)         // [B, 4]
{
  __shared__ float sh_h[16 * HSTR];     // hidden state; last slot = trash can
  __shared__ float sh_gi[GNP * GSTR];   // x@Wih C-tiles, col-major
  __shared__ float sh_gh[GNP * GSTR];   // h@Whh C-tiles, col-major
  __shared__ float sh_bih[GNP];         // bih (n-gate needs it alone)
  __shared__ float sh_bhh[GNP];         // bhh (n-gate needs it alone)
  __shared__ float sh_bs[GNP];          // bih+bhh (r/z gates)
  __shared__ float sh_enc[96];          // [0..49]=enc_in_W, [50..74]=enc_in_b, pad 0

  const int lane = threadIdx.x;         // 0..31 (wave32)
  const int mrow = lane & 15;           // A/C row group
  const int khi  = lane >> 4;           // half-wave selector
  const int ncol = lane & 15;           // B/C column
  const int b0   = blockIdx.x * 16;     // batch tile base

  // ---- stage constants into LDS ----
  for (int i = lane; i < 96; i += 32) {
    float e = 0.0f;
    if (i < 50)      e = enc_in_W[i];
    else if (i < 75) e = enc_in_b[i - 50];
    sh_enc[i] = e;
  }
  for (int i = lane; i < GNP; i += 32) {
    const float vi = (i < GN) ? bih[i] : 0.0f;
    const float vh = (i < GN) ? bhh[i] : 0.0f;
    sh_bih[i] = vi;
    sh_bhh[i] = vh;
    sh_bs[i]  = vi + vh;
  }
  for (int i = lane; i < 16 * HSTR; i += 32) sh_h[i] = 0.0f;

  // ---- weights -> f16 B-matrix layout, register resident (10 x v16h) ----
  v16h wbih[NT], wbhh[NT];
#pragma unroll
  for (int tile = 0; tile < NT; ++tile) {
#pragma unroll
    for (int j = 0; j < 16; ++j) {
      const int k = khi * 16 + j;
      const int c = tile * 16 + ncol;
      const bool ok = (k < RNN) && (c < GN);
      wbih[tile][j] = (_Float16)(ok ? Wih[k * GN + c] : 0.0f);
      wbhh[tile][j] = (_Float16)(ok ? Whh[k * GN + c] : 0.0f);
    }
  }
  __syncthreads();

  const v8f zacc = {0.f, 0.f, 0.f, 0.f, 0.f, 0.f, 0.f, 0.f};

  // ---- reverse-time GRU scan ----
  for (int t = SEQL - 1; t >= 0; --t) {
    // x projection -> A layout; h -> A layout (pad cols of sh_h are zero)
    const float* orow = obs + ((size_t)(b0 + mrow) * SEQL + t) * OBS;
    const float o0 = orow[0];
    const float o1 = orow[1];
    v16h ax, ah;
#pragma unroll
    for (int j = 0; j < 16; ++j) {
      const int k = khi * 16 + j; // 0..31
      const float pre = fmaf(o0, sh_enc[k], fmaf(o1, sh_enc[25 + k], sh_enc[50 + k]));
      const float xv  = (k < RNN) ? tanh_ff(pre) : 0.0f;   // select, no branch
      ax[j] = (_Float16)xv;
      ah[j] = (_Float16)sh_h[mrow * HSTR + k];             // zeros in pad cols
    }

    // 10 WMMAs per step: gi = x_proj@Wih, gh = h@Whh; dump col-major (b128)
#pragma unroll
    for (int tile = 0; tile < NT; ++tile) {
      v8f ci = __builtin_amdgcn_wmma_f32_16x16x32_f16(
          false, ax, false, wbih[tile], (short)0, zacc, false, false);
      v8f ch = __builtin_amdgcn_wmma_f32_16x16x32_f16(
          false, ah, false, wbhh[tile], (short)0, zacc, false, false);
      float* pi = &sh_gi[(tile * 16 + ncol) * GSTR + 8 * khi];
      float* ph = &sh_gh[(tile * 16 + ncol) * GSTR + 8 * khi];
#pragma unroll
      for (int r = 0; r < 8; ++r) { pi[r] = ci[r]; ph[r] = ch[r]; }
    }
    __syncthreads();

    // gate elementwise: 16 rows x 25 cols; 13 full iterations, branch-free.
    // Out-of-range elements (idx>=400 -> j==25) read in-bounds padded data and
    // are steered into a dead LDS slot that is never read back.
#pragma unroll
    for (int it = 0; it < 13; ++it) {
      const int idx = lane + 32 * it;   // 0..415
      const int m = idx & 15;
      const int j = idx >> 4;           // 0..25 (j==25 is the dead tail)
      const float rr = sigm_f(sh_gi[j * GSTR + m] + sh_gh[j * GSTR + m] + sh_bs[j]);
      const float zz = sigm_f(sh_gi[(25 + j) * GSTR + m] +
                              sh_gh[(25 + j) * GSTR + m] + sh_bs[25 + j]);
      const float nn = tanh_ff(sh_gi[(50 + j) * GSTR + m] + sh_bih[50 + j] +
                               rr * (sh_gh[(50 + j) * GSTR + m] + sh_bhh[50 + j]));
      const float hold = sh_h[m * HSTR + j];
      const int   dst  = (idx < 16 * RNN) ? (m * HSTR + j) : (16 * HSTR - 1);
      sh_h[dst] = (1.0f - zz) * nn + zz * hold;
    }
    __syncthreads();
  }

  // ---- stats head: [16,25] @ [25,8]; 128 outputs across 32 lanes ----
#pragma unroll
  for (int q = 0; q < 4; ++q) {
    const int idx = lane * 4 + q;       // 0..127
    const int m = idx >> 3;
    const int c = idx & 7;
    float acc = statsb[c];
    for (int k = 0; k < RNN; ++k)
      acc = fmaf(sh_h[m * HSTR + k], statsW[k * 8 + c], acc);
    const int gb = b0 + m;
    if (c < LAT) {
      out_z0[gb * LAT + c] = acc;       // z0 = mu (sample=False)
      out_mu[gb * LAT + c] = acc;
    } else {
      out_lv[gb * LAT + (c - LAT)] = acc;
    }
  }
}

// =====================================================================
// Kernel 2: dopri5 latent ODE + fused decoder, one thread per batch row.
// Weight rows read contiguously (merge to uniform ds_load_b128 broadcasts).
// =====================================================================
__device__ __forceinline__ void ode_f(const float* __restrict__ zi,
                                      float* __restrict__ ko,
                                      const float* sW1, const float* sB1,
                                      const float* sW2, const float* sB2) {
  float hv[ODEH];
#pragma unroll
  for (int j = 0; j < ODEH; ++j) hv[j] = sB1[j];
#pragma unroll
  for (int i = 0; i < LAT; ++i) {
    const float zv = zi[i];
#pragma unroll
    for (int j = 0; j < ODEH; ++j)               // row-contiguous W1 reads
      hv[j] = fmaf(zv, sW1[i * ODEH + j], hv[j]);
  }
  float acc[LAT];
#pragma unroll
  for (int c = 0; c < LAT; ++c) acc[c] = sB2[c];
#pragma unroll
  for (int j = 0; j < ODEH; ++j) {
    const float e = (hv[j] > 0.0f) ? hv[j] : (exp2_hw(LOG2E * hv[j]) - 1.0f); // ELU
#pragma unroll
    for (int c = 0; c < LAT; ++c)                // row-contiguous W2 reads
      acc[c] = fmaf(e, sW2[j * LAT + c], acc[c]);
  }
#pragma unroll
  for (int c = 0; c < LAT; ++c) ko[c] = acc[c];
}

__global__ __launch_bounds__(32) void ode_decode(
    const float* __restrict__ pred_times, // [T]
    const float* __restrict__ ode_W1, const float* __restrict__ ode_b1,
    const float* __restrict__ ode_W2, const float* __restrict__ ode_b2,
    const float* __restrict__ dec_W1, const float* __restrict__ dec_b1,
    const float* __restrict__ dec_W2, const float* __restrict__ dec_b2,
    const float* __restrict__ z0,         // [B, 4] (written by kernel 1)
    float* __restrict__ z_traj,           // [T, B, 4]
    float* __restrict__ x_pred)           // [T, B, 2]
{
  __shared__ float sW1[LAT * ODEH], sB1[ODEH], sW2[ODEH * LAT], sB2[LAT];
  __shared__ float dW1s[LAT * ODEH], dB1s[ODEH], dW2s[ODEH * OBS], dB2s[OBS];
  __shared__ float sT[TT];

  const int tid = threadIdx.x;
  for (int i = tid; i < LAT * ODEH; i += 32) { sW1[i] = ode_W1[i]; dW1s[i] = dec_W1[i]; }
  for (int i = tid; i < ODEH;       i += 32) { sB1[i] = ode_b1[i]; dB1s[i] = dec_b1[i]; }
  for (int i = tid; i < ODEH * LAT; i += 32) sW2[i] = ode_W2[i];
  for (int i = tid; i < LAT;        i += 32) sB2[i] = ode_b2[i];
  for (int i = tid; i < ODEH * OBS; i += 32) dW2s[i] = dec_W2[i];
  for (int i = tid; i < OBS;        i += 32) dB2s[i] = dec_b2[i];
  for (int i = tid; i < TT;         i += 32) sT[i] = pred_times[i];
  __syncthreads();

  const int b = blockIdx.x * 32 + tid;    // 0..4095

  float z[LAT];
#pragma unroll
  for (int c = 0; c < LAT; ++c) z[c] = z0[b * LAT + c];

  // fused decoder + trajectory store for time index t
  auto emit = [&](const float* zi, int t) {
    float hv[ODEH];
#pragma unroll
    for (int j = 0; j < ODEH; ++j) hv[j] = dB1s[j];
#pragma unroll
    for (int i = 0; i < LAT; ++i) {
      const float zv = zi[i];
#pragma unroll
      for (int j = 0; j < ODEH; ++j)
        hv[j] = fmaf(zv, dW1s[i * ODEH + j], hv[j]);
    }
    float o0 = dB2s[0], o1 = dB2s[1];
#pragma unroll
    for (int j = 0; j < ODEH; ++j) {
      const float e = fmaxf(hv[j], 0.0f);
      o0 = fmaf(e, dW2s[j * OBS + 0], o0);
      o1 = fmaf(e, dW2s[j * OBS + 1], o1);
    }
    const size_t zb = ((size_t)t * BATCH + b) * LAT;
#pragma unroll
    for (int c = 0; c < LAT; ++c) z_traj[zb + c] = zi[c];
    const size_t xb = ((size_t)t * BATCH + b) * OBS;
    x_pred[xb + 0] = o0;
    x_pred[xb + 1] = o1;
  };

  emit(z, 0);

  float k1[LAT], k2[LAT], k3[LAT], k4[LAT], k5[LAT], k6[LAT], zt[LAT];
  for (int s = 0; s < TT - 1; ++s) {
    const float h = sT[s + 1] - sT[s];
    ode_f(z, k1, sW1, sB1, sW2, sB2);
#pragma unroll
    for (int c = 0; c < LAT; ++c) zt[c] = fmaf(h, 0.2f * k1[c], z[c]);
    ode_f(zt, k2, sW1, sB1, sW2, sB2);
#pragma unroll
    for (int c = 0; c < LAT; ++c)
      zt[c] = z[c] + h * (0.075f * k1[c] + 0.225f * k2[c]);
    ode_f(zt, k3, sW1, sB1, sW2, sB2);
#pragma unroll
    for (int c = 0; c < LAT; ++c)
      zt[c] = z[c] + h * ((44.0f / 45.0f) * k1[c] - (56.0f / 15.0f) * k2[c] +
                          (32.0f / 9.0f) * k3[c]);
    ode_f(zt, k4, sW1, sB1, sW2, sB2);
#pragma unroll
    for (int c = 0; c < LAT; ++c)
      zt[c] = z[c] + h * ((19372.0f / 6561.0f) * k1[c] - (25360.0f / 2187.0f) * k2[c] +
                          (64448.0f / 6561.0f) * k3[c] - (212.0f / 729.0f) * k4[c]);
    ode_f(zt, k5, sW1, sB1, sW2, sB2);
#pragma unroll
    for (int c = 0; c < LAT; ++c)
      zt[c] = z[c] + h * ((9017.0f / 3168.0f) * k1[c] - (355.0f / 33.0f) * k2[c] +
                          (46732.0f / 5247.0f) * k3[c] + (49.0f / 176.0f) * k4[c] -
                          (5103.0f / 18656.0f) * k5[c]);
    ode_f(zt, k6, sW1, sB1, sW2, sB2);
#pragma unroll
    for (int c = 0; c < LAT; ++c)
      z[c] = z[c] + h * ((35.0f / 384.0f) * k1[c] + (500.0f / 1113.0f) * k3[c] +
                         (125.0f / 192.0f) * k4[c] - (2187.0f / 6784.0f) * k5[c] +
                         (11.0f / 84.0f) * k6[c]);

    emit(z, s + 1);
  }
}

// =====================================================================
// Launcher
// =====================================================================
extern "C" void kernel_launch(void* const* d_in, const int* in_sizes, int n_in,
                              void* d_out, int out_size, void* d_ws, size_t ws_size,
                              hipStream_t stream) {
  (void)in_sizes; (void)n_in; (void)d_ws; (void)ws_size; (void)out_size;

  const float* obs      = (const float*)d_in[0];
  const float* ptimes   = (const float*)d_in[1];
  const float* enc_in_W = (const float*)d_in[2];
  const float* enc_in_b = (const float*)d_in[3];
  const float* gru_Wih  = (const float*)d_in[4];
  const float* gru_Whh  = (const float*)d_in[5];
  const float* gru_bih  = (const float*)d_in[6];
  const float* gru_bhh  = (const float*)d_in[7];
  const float* statsW   = (const float*)d_in[8];
  const float* statsb   = (const float*)d_in[9];
  const float* ode_W1   = (const float*)d_in[10];
  const float* ode_b1   = (const float*)d_in[11];
  const float* ode_W2   = (const float*)d_in[12];
  const float* ode_b2   = (const float*)d_in[13];
  const float* dec_W1   = (const float*)d_in[14];
  const float* dec_b1   = (const float*)d_in[15];
  const float* dec_W2   = (const float*)d_in[16];
  const float* dec_b2   = (const float*)d_in[17];

  float* out    = (float*)d_out;
  float* x_pred = out;                                   // [T,B,OBS]
  float* z_traj = out + (size_t)TT * BATCH * OBS;        // [T,B,LAT]
  float* z0     = z_traj + (size_t)TT * BATCH * LAT;     // [B,LAT]
  float* mu     = z0 + (size_t)BATCH * LAT;              // [B,LAT]
  float* lv     = mu + (size_t)BATCH * LAT;              // [B,LAT]

  gru_encoder_wmma<<<BATCH / 16, 32, 0, stream>>>(
      obs, enc_in_W, enc_in_b, gru_Wih, gru_Whh, gru_bih, gru_bhh,
      statsW, statsb, z0, mu, lv);

  ode_decode<<<BATCH / 32, 32, 0, stream>>>(
      ptimes, ode_W1, ode_b1, ode_W2, ode_b2,
      dec_W1, dec_b1, dec_W2, dec_b2, z0, z_traj, x_pred);
}
